// MultiHeadSelfAttention_70720931496387
// MI455X (gfx1250) — matrix operand order repaired and checked by reading, not compiled, not run on hardware
//
#include <hip/hip_runtime.h>

#define E_DIM 1024
#define NH    16
#define HD    64
#define SEQ   2048
#define BATCH 2
#define MTOT  (BATCH * SEQ)   // 4096

typedef __attribute__((ext_vector_type(16))) __bf16 v16bf;
typedef __attribute__((ext_vector_type(8)))  float  v8f;

union ABFrag { unsigned int u[8]; v16bf v; };

// ---- async global->LDS (CDNA5), with portable fallback --------------------
#if __has_builtin(__builtin_amdgcn_global_load_async_to_lds_b64) && \
    __has_builtin(__builtin_amdgcn_global_load_async_to_lds_b128)
#define HAVE_ASYNC_LDS 1
#else
#define HAVE_ASYNC_LDS 0
#endif

#if HAVE_ASYNC_LDS
typedef int v2i __attribute__((vector_size(8)));
typedef int v4i __attribute__((vector_size(16)));
typedef __attribute__((address_space(1))) v2i* as1_v2i;
typedef __attribute__((address_space(3))) v2i* as3_v2i;
typedef __attribute__((address_space(1))) v4i* as1_v4i;
typedef __attribute__((address_space(3))) v4i* as3_v4i;

__device__ __forceinline__ void async_g2l_b64(const void* g, void* l) {
    __builtin_amdgcn_global_load_async_to_lds_b64(
        (as1_v2i)(v2i*)g, (as3_v2i)(v2i*)l, 0, 0);
}
__device__ __forceinline__ void async_g2l_b128(const void* g, void* l) {
    __builtin_amdgcn_global_load_async_to_lds_b128(
        (as1_v4i)(v4i*)g, (as3_v4i)(v4i*)l, 0, 0);
}
__device__ __forceinline__ void async_wait0() {
    asm volatile("s_wait_asynccnt 0x0" ::: "memory");
}
#endif

__device__ __forceinline__ unsigned int f2bf1(float f) {
    unsigned int x = __float_as_uint(f);
    x += 0x7FFFu + ((x >> 16) & 1u);   // round to nearest even
    return x >> 16;
}
__device__ __forceinline__ unsigned int pack2(float a, float b) {
    return f2bf1(a) | (f2bf1(b) << 16);
}

// A-fragment (16-bit 16x32): lane-half g picks K sub-ranges; dword pairs are
// contiguous: {4g..4g+3} then {8+4g..8+4g+3}.  B-fragment: same as A.
__device__ __forceinline__ void frag_a_lds(ABFrag& f, const unsigned int* row, int g) {
    *(uint4*)&f.u[0] = *(const uint4*)(row + 4 * g);
    *(uint4*)&f.u[4] = *(const uint4*)(row + 8 + 4 * g);
}
__device__ __forceinline__ void frag_b_lds(ABFrag& f, const unsigned int* row, int g, int koff) {
    *(uint4*)&f.u[0] = *(const uint4*)(row + koff + 4 * g);
    *(uint4*)&f.u[4] = *(const uint4*)(row + koff + 8 + 4 * g);
}

// ---------------------------------------------------------------------------
// Kernel 1: qkv = x @ W_qkv^T + b_qkv, scattered to Q/K/V [B,H,S,D] bf16
// ---------------------------------------------------------------------------
__global__ __launch_bounds__(256) void qkv_gemm_kernel(
    const float* __restrict__ x, const float* __restrict__ Wqkv,
    const float* __restrict__ bqkv,
    unsigned short* __restrict__ Qb, unsigned short* __restrict__ Kb,
    unsigned short* __restrict__ Vb)
{
    __shared__ __align__(16) unsigned int lA[128][20];   // stride 80B: 16B-aligned rows
    __shared__ __align__(16) unsigned int lB[64][20];

    const int t = threadIdx.x;
    const int w = t >> 5, lane = t & 31;
    const int g = lane >> 4, ln = lane & 15;
    const int row0 = blockIdx.x * 128;
    const int col0 = blockIdx.y * 64;

    v8f acc[4] = {};

    for (int k0 = 0; k0 < E_DIM; k0 += 32) {
        if (k0 + 32 < E_DIM)
            __builtin_prefetch(x + (row0 + (t >> 4)) * E_DIM + k0 + 32, 0, 0);
#pragma unroll
        for (int i = 0; i < 8; ++i) {       // A: fp32 -> bf16 convert stage
            int idx = t + i * 256;
            int r = idx >> 4, p = idx & 15;
            const float* src = x + (size_t)(row0 + r) * E_DIM + k0 + 2 * p;
            lA[r][p] = pack2(src[0], src[1]);
        }
#pragma unroll
        for (int i = 0; i < 4; ++i) {       // B: fp32 -> bf16 convert stage
            int idx = t + i * 256;
            int r = idx >> 4, p = idx & 15;
            const float* src = Wqkv + (size_t)(col0 + r) * E_DIM + k0 + 2 * p;
            lB[r][p] = pack2(src[0], src[1]);
        }
        __syncthreads();

        ABFrag a;
        frag_a_lds(a, lA[w * 16 + ln], g);
        ABFrag bf[4];
#pragma unroll
        for (int j = 0; j < 4; ++j) frag_b_lds(bf[j], lB[16 * j + ln], g, 0);

#pragma unroll
        for (int j = 0; j < 4; ++j)
            acc[j] = __builtin_amdgcn_wmma_f32_16x16x32_bf16(
                false, a.v, false, bf[j].v, (short)0, acc[j], false, false);
        __syncthreads();
    }

#pragma unroll
    for (int j = 0; j < 4; ++j) {
#pragma unroll
        for (int r = 0; r < 8; ++r) {
            int grow = row0 + w * 16 + r + 8 * g;     // C/D row = r + 8*half
            int f    = col0 + 16 * j + ln;            // C/D col = lane
            float val = acc[j][r] + bqkv[f];
            int c   = f >> 10;
            int rem = f & 1023;
            int h = rem >> 6, d = rem & 63;
            int b = grow >> 11, s = grow & 2047;
            unsigned short* dst = (c == 0) ? Qb : (c == 1) ? Kb : Vb;
            dst[(((size_t)(b * NH + h) * SEQ) + s) * HD + d] =
                (unsigned short)f2bf1(val);
        }
    }
}

// ---------------------------------------------------------------------------
// Kernel 2: flash-style attention. Block = (b, h, 128 q rows), wave = 16 rows.
// ---------------------------------------------------------------------------
__global__ __launch_bounds__(256) void attn_kernel(
    const unsigned short* __restrict__ Qb,
    const unsigned short* __restrict__ Kb,
    const unsigned short* __restrict__ Vb,
    unsigned short* __restrict__ Ob)
{
    __shared__ __align__(16) unsigned int lK[32][36];        // 144B rows
    __shared__ __align__(16) unsigned int lVt[64][20];       // transposed V
    __shared__ __align__(16) unsigned short lP[8][16][40];   // 80B rows

    const int t = threadIdx.x;
    const int w = t >> 5, lane = t & 31;
    const int g = lane >> 4, ln = lane & 15;
    const int b = blockIdx.z, h = blockIdx.y;
    const int q0 = blockIdx.x * 128 + w * 16;

    const size_t head = ((size_t)(b * NH + h)) * SEQ * HD;
    const unsigned short* Kbase = Kb + head;
    const unsigned short* Vbase = Vb + head;

    // Q fragments: two K-halves of D=64, 4x global_load_b128, live in regs
    ABFrag qa0, qa1;
    {
        const unsigned short* qr = Qb + head + (size_t)(q0 + ln) * HD;
        *(uint4*)&qa0.u[0] = *(const uint4*)(qr + 8 * g);
        *(uint4*)&qa0.u[4] = *(const uint4*)(qr + 16 + 8 * g);
        *(uint4*)&qa1.u[0] = *(const uint4*)(qr + 32 + 8 * g);
        *(uint4*)&qa1.u[4] = *(const uint4*)(qr + 48 + 8 * g);
    }

    v8f o[4] = {};
    float mrow[8], lrow[8];
#pragma unroll
    for (int r = 0; r < 8; ++r) { mrow[r] = -1e30f; lrow[r] = 0.f; }

    for (int kv0 = 0; kv0 < SEQ; kv0 += 32) {
        // stage K tile (bf16 straight copy): async DMA to LDS when available
#if HAVE_ASYNC_LDS
        {
            int r = t >> 3, p = t & 7;   // 32 rows x 8 qwords
            async_g2l_b64(Kbase + (size_t)(kv0 + r) * HD + 4 * p, &lK[r][2 * p]);
        }
#else
#pragma unroll
        for (int i = 0; i < 4; ++i) {
            int idx = t + i * 256;
            int r = idx >> 5, p = idx & 31;
            lK[r][p] = *(const unsigned int*)(Kbase + (size_t)(kv0 + r) * HD + 2 * p);
        }
#endif
        // stage V transposed: lVt[d][p] = {V[kv0+2p][d], V[kv0+2p+1][d]}
#pragma unroll
        for (int i = 0; i < 4; ++i) {
            int idx = t + i * 256;
            int d = idx >> 4, p = idx & 15;
            unsigned int v0 = Vbase[(size_t)(kv0 + 2 * p) * HD + d];
            unsigned int v1 = Vbase[(size_t)(kv0 + 2 * p + 1) * HD + d];
            lVt[d][p] = v0 | (v1 << 16);
        }
#if HAVE_ASYNC_LDS
        async_wait0();
#endif
        __syncthreads();

        // S = Q K^T for two 16-col kv sub-tiles, K-dim = D = 64 (2 WMMAs each)
        v8f s[2] = {};
#pragma unroll
        for (int jn = 0; jn < 2; ++jn) {
            ABFrag bk0, bk1;
            frag_b_lds(bk0, lK[16 * jn + ln], g, 0);
            frag_b_lds(bk1, lK[16 * jn + ln], g, 16);
            s[jn] = __builtin_amdgcn_wmma_f32_16x16x32_bf16(
                false, qa0.v, false, bk0.v, (short)0, s[jn], false, false);
            s[jn] = __builtin_amdgcn_wmma_f32_16x16x32_bf16(
                false, qa1.v, false, bk1.v, (short)0, s[jn], false, false);
        }

        // online softmax over the 16x32 score tile (rows r+8g, 16 lanes/row)
        const float scale = 0.125f;   // 1/sqrt(64)
#pragma unroll
        for (int r = 0; r < 8; ++r) {
            float a0 = s[0][r] * scale, a1 = s[1][r] * scale;
            float lm = fmaxf(a0, a1);
#pragma unroll
            for (int msk = 1; msk < 16; msk <<= 1)
                lm = fmaxf(lm, __shfl_xor(lm, msk, 32));
            float nm = fmaxf(mrow[r], lm);
            float alpha = __expf(mrow[r] - nm);
            float p0 = __expf(a0 - nm);
            float p1 = __expf(a1 - nm);
            float rs = p0 + p1;
#pragma unroll
            for (int msk = 1; msk < 16; msk <<= 1)
                rs += __shfl_xor(rs, msk, 32);
            lrow[r] = lrow[r] * alpha + rs;
            mrow[r] = nm;
#pragma unroll
            for (int dj = 0; dj < 4; ++dj) o[dj][r] *= alpha;
            // P tile to LDS (C-layout -> memory) for A-fragment reread
            lP[w][r + 8 * g][ln]      = (unsigned short)f2bf1(p0);
            lP[w][r + 8 * g][16 + ln] = (unsigned short)f2bf1(p1);
        }
        __syncthreads();

        // O += P (16x32) x V (32x64): 4 WMMAs over d tiles
        ABFrag pa;
        *(uint4*)&pa.u[0] = *(const uint4*)&lP[w][ln][8 * g];
        *(uint4*)&pa.u[4] = *(const uint4*)&lP[w][ln][16 + 8 * g];
#pragma unroll
        for (int dj = 0; dj < 4; ++dj) {
            ABFrag bv;
            frag_b_lds(bv, lVt[16 * dj + ln], g, 0);
            o[dj] = __builtin_amdgcn_wmma_f32_16x16x32_bf16(
                false, pa.v, false, bv.v, (short)0, o[dj], false, false);
        }
        __syncthreads();
    }

    // normalize + store to attn_out [B,S,E] bf16
#pragma unroll
    for (int dj = 0; dj < 4; ++dj) {
#pragma unroll
        for (int r = 0; r < 8; ++r) {
            int q = q0 + r + 8 * g;
            float val = o[dj][r] / lrow[r];
            Ob[(size_t)(b * SEQ + q) * E_DIM + h * HD + 16 * dj + ln] =
                (unsigned short)f2bf1(val);
        }
    }
}

// ---------------------------------------------------------------------------
// Kernel 3: y = attn_out @ W_proj^T + b_proj + residual   (fp32 out)
// ---------------------------------------------------------------------------
__global__ __launch_bounds__(256) void proj_gemm_kernel(
    const unsigned short* __restrict__ Ab,   // bf16 [4096,1024]
    const float* __restrict__ Wproj,
    const float* __restrict__ bproj,
    const float* __restrict__ x,             // residual
    float* __restrict__ y)
{
    __shared__ __align__(16) unsigned int lA[128][20];
    __shared__ __align__(16) unsigned int lB[64][20];

    const int t = threadIdx.x;
    const int w = t >> 5, lane = t & 31;
    const int g = lane >> 4, ln = lane & 15;
    const int row0 = blockIdx.x * 128;
    const int col0 = blockIdx.y * 64;

    v8f acc[4] = {};

    for (int k0 = 0; k0 < E_DIM; k0 += 32) {
        // A is already bf16: async DMA tile copy (128 rows x 4 x b128)
#if HAVE_ASYNC_LDS
#pragma unroll
        for (int i = 0; i < 2; ++i) {
            int idx = t + i * 256;
            int r = idx >> 2, q = idx & 3;
            async_g2l_b128(Ab + (size_t)(row0 + r) * E_DIM + k0 + 8 * q,
                           &lA[r][4 * q]);
        }
#else
#pragma unroll
        for (int i = 0; i < 8; ++i) {
            int idx = t + i * 256;
            int r = idx >> 4, p = idx & 15;
            lA[r][p] = *(const unsigned int*)(Ab + (size_t)(row0 + r) * E_DIM + k0 + 2 * p);
        }
#endif
#pragma unroll
        for (int i = 0; i < 4; ++i) {        // B: fp32 -> bf16 convert stage
            int idx = t + i * 256;
            int r = idx >> 4, p = idx & 15;
            const float* src = Wproj + (size_t)(col0 + r) * E_DIM + k0 + 2 * p;
            lB[r][p] = pack2(src[0], src[1]);
        }
#if HAVE_ASYNC_LDS
        async_wait0();
#endif
        __syncthreads();

        ABFrag a;
        frag_a_lds(a, lA[w * 16 + ln], g);
        ABFrag bf[4];
#pragma unroll
        for (int j = 0; j < 4; ++j) frag_b_lds(bf[j], lB[16 * j + ln], g, 0);
#pragma unroll
        for (int j = 0; j < 4; ++j)
            acc[j] = __builtin_amdgcn_wmma_f32_16x16x32_bf16(
                false, a.v, false, bf[j].v, (short)0, acc[j], false, false);
        __syncthreads();
    }

#pragma unroll
    for (int j = 0; j < 4; ++j) {
#pragma unroll
        for (int r = 0; r < 8; ++r) {
            int grow = row0 + w * 16 + r + 8 * g;
            int f    = col0 + 16 * j + ln;
            size_t off = (size_t)grow * E_DIM + f;
            y[off] = acc[j][r] + bproj[f] + x[off];
        }
    }
}

// ---------------------------------------------------------------------------
// Kernel 4: LayerNorm per row (E=1024), eps=1e-5
// ---------------------------------------------------------------------------
__global__ __launch_bounds__(256) void ln_kernel(
    const float* __restrict__ y, const float* __restrict__ gamma,
    const float* __restrict__ beta, float* __restrict__ out)
{
    __shared__ float red[256];
    const int row = blockIdx.x;
    const int t = threadIdx.x;
    const float* yr = y + (size_t)row * E_DIM;

    float vals[4], s = 0.f, ss = 0.f;
#pragma unroll
    for (int i = 0; i < 4; ++i) {
        float v = yr[t + i * 256];
        vals[i] = v; s += v; ss += v * v;
    }
    red[t] = s; __syncthreads();
    for (int off = 128; off > 0; off >>= 1) {
        if (t < off) red[t] += red[t + off];
        __syncthreads();
    }
    float mean = red[0] * (1.f / E_DIM);
    __syncthreads();
    red[t] = ss; __syncthreads();
    for (int off = 128; off > 0; off >>= 1) {
        if (t < off) red[t] += red[t + off];
        __syncthreads();
    }
    float var = red[0] * (1.f / E_DIM) - mean * mean;
    float inv = rsqrtf(var + 1e-5f);
#pragma unroll
    for (int i = 0; i < 4; ++i) {
        int c = t + i * 256;
        out[(size_t)row * E_DIM + c] = (vals[i] - mean) * inv * gamma[c] + beta[c];
    }
}

// ---------------------------------------------------------------------------
extern "C" void kernel_launch(void* const* d_in, const int* in_sizes, int n_in,
                              void* d_out, int out_size, void* d_ws, size_t ws_size,
                              hipStream_t stream)
{
    const float* x     = (const float*)d_in[0];
    const float* Wqkv  = (const float*)d_in[1];
    const float* bqkv  = (const float*)d_in[2];
    const float* Wproj = (const float*)d_in[3];
    const float* bproj = (const float*)d_in[4];
    const float* gamma = (const float*)d_in[5];
    const float* beta  = (const float*)d_in[6];
    float* out = (float*)d_out;

    char* ws = (char*)d_ws;
    const size_t QSZ = (size_t)BATCH * NH * SEQ * HD * sizeof(unsigned short); // 8 MB
    unsigned short* Qb = (unsigned short*)(ws);
    unsigned short* Kb = (unsigned short*)(ws + QSZ);
    unsigned short* Vb = (unsigned short*)(ws + 2 * QSZ);
    unsigned short* Ob = (unsigned short*)(ws + 3 * QSZ);
    float* y = (float*)ws;   // reuses Q+K region (dead after attention)

    qkv_gemm_kernel<<<dim3(MTOT / 128, (3 * E_DIM) / 64), 256, 0, stream>>>(
        x, Wqkv, bqkv, Qb, Kb, Vb);
    attn_kernel<<<dim3(SEQ / 128, NH, BATCH), 256, 0, stream>>>(Qb, Kb, Vb, Ob);
    proj_gemm_kernel<<<dim3(MTOT / 128, E_DIM / 64), 256, 0, stream>>>(
        Ob, Wproj, bproj, x, y);
    ln_kernel<<<MTOT, 256, 0, stream>>>(y, gamma, beta, out);
}
